// Gate_74371653697964
// MI455X (gfx1250) — 8-bit matrix operand order repaired and checked by reading, compiled, not run on hardware
//
#include <hip/hip_runtime.h>
#include <hip/hip_bf16.h>
#include <stdint.h>

#define DIM 4096
#define NEXP 8
#define TOKS_PER_BLK 16
#define ROW_DW 1026

typedef int v8i __attribute__((ext_vector_type(8)));

__device__ __forceinline__ int quant_i8(float v, float qs) {
    int q = (int)rintf(v * qs);
    q = q < -128 ? -128 : q;
    q = q > 127 ? 127 : q;
    return q;
}

__global__ __launch_bounds__(256) void WeightQuant_kernel(const float* __restrict__ W,
                                                          int8_t* __restrict__ wq,
                                                          float* __restrict__ wsinv_out) {
    __shared__ float red[256];
    const int tid = threadIdx.x;
    float s = 0.f;
    #pragma unroll 8
    for (int i = 0; i < 128; ++i) s += fabsf(W[tid + 256 * i]);
    red[tid] = s;
    __syncthreads();
    for (int off = 128; off > 0; off >>= 1) {
        if (tid < off) red[tid] += red[tid + off];
        __syncthreads();
    }
    const float meanabs = red[0] * (1.0f / 32768.0f);
    const float wsinv = fmaxf(meanabs, 1e-5f);
    const float wscale = 1.0f / wsinv;
    #pragma unroll 8
    for (int i = 0; i < 128; ++i) {
        const int idx = tid + 256 * i;
        int q = (int)rintf(W[idx] * wscale);
        q = q < -1 ? -1 : (q > 1 ? 1 : q);
        wq[idx] = (int8_t)q;
        wq[NEXP * DIM + idx] = 0;
    }
    if (tid == 0) *wsinv_out = wsinv;
}

__global__ __launch_bounds__(256) void Gate_74371653697964_kernel(
    const float* __restrict__ x, const float* __restrict__ bias,
    const int8_t* __restrict__ wq, const float* __restrict__ wsinv_p,
    float* __restrict__ out) {

    __shared__ uint32_t xq[TOKS_PER_BLK * ROW_DW];
    __shared__ float ascale_s[TOKS_PER_BLK];
    __shared__ int partial[8 * 256];

    const int tid  = threadIdx.x;
    const int lane = tid & 31;
    const int wave = tid >> 5;
    const long tok0 = (long)blockIdx.x * TOKS_PER_BLK;

    const int tok = tid >> 4;
    const int j   = tid & 15;
    const float4* src = (const float4*)(x + (tok0 + tok) * DIM) + j;

    float ss = 0.f, amax = 0.f;
    #pragma unroll 8
    for (int i = 0; i < 64; ++i) {
        const float4 v = src[i * 16];
        ss += v.x * v.x + v.y * v.y + v.z * v.z + v.w * v.w;
        amax = fmaxf(amax, fmaxf(fmaxf(fabsf(v.x), fabsf(v.y)),
                                 fmaxf(fabsf(v.z), fabsf(v.w))));
    }
    #pragma unroll
    for (int off = 8; off > 0; off >>= 1) {
        ss   += __shfl_xor(ss, off, 16);
        amax  = fmaxf(amax, __shfl_xor(amax, off, 16));
    }
    float nrm = fmaxf(sqrtf(ss), 1e-12f);
    const float rnorm  = 64.0f / nrm;
    const float ascale = 127.0f / fmaxf(amax * rnorm, 1e-5f);
    const float qs     = rnorm * ascale;
    if (j == 0) ascale_s[tok] = ascale;

    uint32_t* row = xq + tok * ROW_DW;
    #pragma unroll 4
    for (int i = 0; i < 64; ++i) {
        const float4 v = src[i * 16];
        const uint32_t p = (uint32_t)(quant_i8(v.x, qs) & 0xff)
                         | ((uint32_t)(quant_i8(v.y, qs) & 0xff) << 8)
                         | ((uint32_t)(quant_i8(v.z, qs) & 0xff) << 16)
                         | ((uint32_t)(quant_i8(v.w, qs) & 0xff) << 24);
        row[j + 16 * i] = p;
    }
    __syncthreads();

    const int m    = lane & 15;
    const int half = lane >> 4;
    v8i acc = {};
    const uint32_t* arow = xq + m * ROW_DW;
    const uint32_t* brow = (const uint32_t*)(wq + m * DIM);

    #pragma unroll
    for (int c = 0; c < 8; ++c) {
        const int kb = (wave * 8 + c) * 16;
        const uint32_t* ap = arow + kb + half * 4;
        v8i A, B;
        A[0] = ap[0];  A[1] = ap[1];  A[2] = ap[2];  A[3] = ap[3];
        A[4] = ap[8];  A[5] = ap[9];  A[6] = ap[10]; A[7] = ap[11];
        const uint32_t* bp = brow + kb + half * 4;
        B[0] = bp[0];  B[1] = bp[1];  B[2] = bp[2];  B[3] = bp[3];
        B[4] = bp[8];  B[5] = bp[9];  B[6] = bp[10]; B[7] = bp[11];
        acc = __builtin_amdgcn_wmma_i32_16x16x64_iu8(true, A, true, B, acc,
                                                     false, false);
    }
    int* pw = partial + wave * 256;
    #pragma unroll
    for (int r = 0; r < 8; ++r) pw[(r + 8 * half) * 16 + m] = acc[r];
    __syncthreads();

    const int t = tid >> 4;
    const int e = tid & 15;
    int isum = 0;
    #pragma unroll
    for (int w = 0; w < 8; ++w) isum += partial[w * 256 + t * 16 + e];

    const float wsinv = *wsinv_p;
    float logit = (e < NEXP)
        ? ((float)isum * wsinv / ascale_s[t] + bias[e])
        : -__builtin_inff();

    float mx = logit;
    #pragma unroll
    for (int off = 8; off > 0; off >>= 1) mx = fmaxf(mx, __shfl_xor(mx, off, 16));
    const float ex = (e < NEXP) ? expf(logit - mx) : 0.f;
    float sm = ex;
    #pragma unroll
    for (int off = 8; off > 0; off >>= 1) sm += __shfl_xor(sm, off, 16);

    if (e < NEXP) out[(tok0 + t) * NEXP + e] = ex / sm;
}

extern "C" void kernel_launch(void* const* d_in, const int* in_sizes, int n_in,
                              void* d_out, int out_size, void* d_ws, size_t ws_size,
                              hipStream_t stream) {
    const float* x = (const float*)d_in[0];
    const float* W = (const float*)d_in[1];
    const float* b = (const float*)d_in[2];
    float* out = (float*)d_out;

    int8_t* wq    = (int8_t*)d_ws;
    float*  wsinv = (float*)((char*)d_ws + 16 * DIM);

    const int n_tok  = in_sizes[0] / DIM;
    const int blocks = n_tok / TOKS_PER_BLK;

    WeightQuant_kernel<<<1, 256, 0, stream>>>(W, wq, wsinv);
    Gate_74371653697964_kernel<<<blocks, 256, 0, stream>>>(x, b, wq, wsinv, out);
}
